// _NonLocalBlockND_30717606101447
// MI455X (gfx1250) — matrix operand order repaired and checked by reading, not compiled, not run on hardware
//
#include <hip/hip_runtime.h>

typedef __bf16 bf16;
typedef __attribute__((ext_vector_type(16))) __bf16 v16bf;
typedef __attribute__((ext_vector_type(8)))  __bf16 v8bf;
typedef __attribute__((ext_vector_type(8)))  float  v8f;

#define BB 8
#define CC 256
#define NN 2304
#define WAVES 4
#define NJ (NN / 32)

// K tile rows padded: 256 + 8 halves = 528 B ; V tile rows: 32 + 8 halves = 80 B
#define KPITCH 264
#define VPITCH 40

// ---------------------------------------------------------------------------
// Pre-pass: x (f32, b,C,N) -> Xb bf16 (b,C,N) and Xt bf16 (b,N,C)
// ---------------------------------------------------------------------------
__global__ void cvt_x_kernel(const float* __restrict__ x,
                             bf16* __restrict__ Xb, bf16* __restrict__ Xt) {
  __shared__ __align__(16) bf16 tile[32][34];
  const int nt = blockIdx.x;
  const int ct = blockIdx.y;
  const int b  = blockIdx.z;
  const int tx = threadIdx.x & 31;
  const int ty = threadIdx.x >> 5;
  const size_t basein = ((size_t)b * CC + ct * 32) * NN + nt * 32;
#pragma unroll
  for (int i = 0; i < 4; ++i) {
    const int r = ty + i * 8;
    const float v = x[basein + (size_t)r * NN + tx];
    const bf16 h = (bf16)v;
    Xb[basein + (size_t)r * NN + tx] = h;
    tile[tx][r] = h;
  }
  __syncthreads();
  const size_t baseout = ((size_t)b * NN + nt * 32) * CC + ct * 32;
#pragma unroll
  for (int i = 0; i < 4; ++i) {
    const int r = ty + i * 8;
    Xt[baseout + (size_t)r * CC + tx] = tile[r][tx];
  }
}

__global__ void cvt_w_kernel(const float* __restrict__ w, bf16* __restrict__ wb) {
  const int i = blockIdx.x * 256 + threadIdx.x;
  wb[i] = (bf16)w[i];
}

// ---------------------------------------------------------------------------
// CDNA5 async global->LDS copy (ASYNCcnt path) + waits
// ---------------------------------------------------------------------------
__device__ __forceinline__ void async_b128(unsigned lds_off, const bf16* g) {
  asm volatile("global_load_async_to_lds_b128 %0, %1, off"
               :: "v"(lds_off), "v"(g) : "memory");
}
__device__ __forceinline__ void wait_async_le16() {
  asm volatile("s_wait_asynccnt 0x10" ::: "memory");
}
__device__ __forceinline__ void wait_async_0() {
  asm volatile("s_wait_asynccnt 0x0" ::: "memory");
}
__device__ __forceinline__ unsigned lds_off_u32(const void* p) {
  // shared-aperture flat address keeps LDS byte offset in low 32 bits
  return (unsigned)(size_t)p;
}
__device__ __forceinline__ v16bf ldsv16(const bf16* p) {
  return *(const v16bf*)p;
}

// ---------------------------------------------------------------------------
// 16-lane-group reductions (rows live in lane halves of wave32)
// ---------------------------------------------------------------------------
__device__ __forceinline__ float redmax16(float v) {
  v = fmaxf(v, __shfl_xor(v, 1, 32));
  v = fmaxf(v, __shfl_xor(v, 2, 32));
  v = fmaxf(v, __shfl_xor(v, 4, 32));
  v = fmaxf(v, __shfl_xor(v, 8, 32));
  return v;
}
__device__ __forceinline__ float redsum16(float v) {
  v += __shfl_xor(v, 1, 32);
  v += __shfl_xor(v, 2, 32);
  v += __shfl_xor(v, 4, 32);
  v += __shfl_xor(v, 8, 32);
  return v;
}

// ---------------------------------------------------------------------------
// Fused flash-softmax attention with async double-buffered K/V staging.
// One wave per 16-row query tile; K/V tiles shared by the 4 waves via LDS.
// ---------------------------------------------------------------------------
__global__ __launch_bounds__(WAVES * 32, 1) void attn_kernel(
    const bf16* __restrict__ Xt, const bf16* __restrict__ Xb,
    bf16* __restrict__ Yt) {
  __shared__ __align__(16) bf16 Kls[2][32][KPITCH];   // keys: 32 rows x 256 c
  __shared__ __align__(16) bf16 Vls[2][256][VPITCH];  // vals: 256 c-rows x 32 n
  __shared__ __align__(16) bf16 pshm[WAVES][16 * 32];

  const int t    = threadIdx.x;
  const int lane = t & 31;
  const int wave = t >> 5;
  const int b    = blockIdx.y;
  const int row0 = blockIdx.x * (WAVES * 16) + wave * 16;
  const int l15  = lane & 15;
  const int hsel = lane >> 4;
  const int hoff = hsel * 8;
  const int boff = hsel * 16;

  const bf16* XtB = Xt + (size_t)b * NN * CC;
  const bf16* XbB = Xb + (size_t)b * CC * NN;

  // per-thread async-copy chunk coordinates (16-byte chunks)
  const int krow0 = t >> 5, kcol = t & 31;   // K: row krow0+4i, 16B col kcol
  const int vrow0 = t >> 2, vcol = t & 3;    // V: row vrow0+32i, 16B col vcol
  const bf16* kg0 = XtB + (size_t)krow0 * CC + kcol * 8;
  const bf16* vg0 = XbB + (size_t)vrow0 * NN + vcol * 8;
  unsigned kls_b[2], vls_b[2];
#pragma unroll
  for (int u = 0; u < 2; ++u) {
    kls_b[u] = lds_off_u32(&Kls[u][krow0][kcol * 8]);
    vls_b[u] = lds_off_u32(&Vls[u][vrow0][vcol * 8]);
  }

  auto issue = [&](int buf, int jc) {
    const bf16* kg = kg0 + (size_t)jc * CC;
    const bf16* vg = vg0 + jc;
#pragma unroll
    for (int i = 0; i < 8; ++i)
      async_b128(kls_b[buf] + i * 4 * KPITCH * 2, kg + (size_t)i * 4 * CC);
#pragma unroll
    for (int i = 0; i < 8; ++i)
      async_b128(vls_b[buf] + i * 32 * VPITCH * 2, vg + (size_t)i * 32 * NN);
  };

  // ---- Q fragments (A layout), cached in VGPRs, reused for all 72 blocks
  v16bf qa[8];
  {
    const bf16* qbase = XtB + (size_t)(row0 + l15) * CC;
#pragma unroll
    for (int kt = 0; kt < 8; ++kt) {
      const bf16* p = qbase + kt * 32 + boff;
      v8bf lo = *(const v8bf*)(p);
      v8bf hi = *(const v8bf*)(p + 8);
#pragma unroll
      for (int e = 0; e < 8; ++e) { qa[kt][e] = lo[e]; qa[kt][e + 8] = hi[e]; }
    }
  }

  v8f acc[16];
#pragma unroll
  for (int i = 0; i < 16; ++i) acc[i] = (v8f){};
  float mrun[8], lrun[8];
#pragma unroll
  for (int v = 0; v < 8; ++v) { mrun[v] = -3.0e30f; lrun[v] = 0.0f; }

  bf16* pbuf = &pshm[wave][0];

  issue(0, 0);
  for (int jj = 0; jj < NJ; ++jj) {
    const int buf = jj & 1;
    if (jj + 1 < NJ) {
      issue(buf ^ 1, (jj + 1) * 32);  // prefetch next block (overlaps compute)
      wait_async_le16();              // own copies for current block done
    } else {
      wait_async_0();
    }
    __syncthreads();                  // whole tile resident for all waves

    // ---- S = Q K^T for 32 keys (two 16-col D tiles), K-dim = C = 256
    // manually double-buffered DS fragment loads
    const bf16* k0p = &Kls[buf][l15][boff];
    const bf16* k1p = &Kls[buf][16 + l15][boff];
    v16bf kb0 = ldsv16(k0p);
    v16bf kb1 = ldsv16(k1p);
    v8f s0 = (v8f){}, s1 = (v8f){};
#pragma unroll
    for (int kt = 0; kt < 8; ++kt) {
      const int nk = (kt + 1) & 7;    // wrap-around dummy load on last iter
      v16bf nb0 = ldsv16(k0p + nk * 32);
      v16bf nb1 = ldsv16(k1p + nk * 32);
      s0 = __builtin_amdgcn_wmma_f32_16x16x32_bf16(false, qa[kt], false, kb0,
                                                   (short)0, s0, false, false);
      s1 = __builtin_amdgcn_wmma_f32_16x16x32_bf16(false, qa[kt], false, kb1,
                                                   (short)0, s1, false, false);
      kb0 = nb0; kb1 = nb1;
    }

    // ---- online softmax (rows = v + 8*hsel, cols across 16 lanes)
    v8f svec;
#pragma unroll
    for (int v = 0; v < 8; ++v) {
      const float bm   = redmax16(fmaxf(s0[v], s1[v]));
      const float mnew = fmaxf(mrun[v], bm);
      const float cs   = __expf(mrun[v] - mnew);
      const float p0   = __expf(s0[v] - mnew);
      const float p1   = __expf(s1[v] - mnew);
      s0[v] = p0; s1[v] = p1;
      lrun[v] = lrun[v] * cs + redsum16(p0 + p1);
      mrun[v] = mnew;
      svec[v] = cs;
    }
#pragma unroll
    for (int ctv = 0; ctv < 16; ++ctv) acc[ctv] *= svec;

    // ---- P: D layout -> LDS -> A layout (per-wave buffer, in-order DS)
#pragma unroll
    for (int v = 0; v < 8; ++v) {
      const int m = v + hoff;
      pbuf[m * 32 + l15]      = (bf16)s0[v];
      pbuf[m * 32 + 16 + l15] = (bf16)s1[v];
    }
    v16bf pA;
    {
      const bf16* p = pbuf + l15 * 32 + boff;
      v8bf lo = *(const v8bf*)(p);
      v8bf hi = *(const v8bf*)(p + 8);
#pragma unroll
      for (int e = 0; e < 8; ++e) { pA[e] = lo[e]; pA[e + 8] = hi[e]; }
    }

    // ---- acc += P @ V from LDS (double-buffered fragment loads)
    const bf16* vp = &Vls[buf][l15][boff];
    v16bf vf = ldsv16(vp);
#pragma unroll
    for (int ct = 0; ct < 16; ++ct) {
      v16bf vn = ldsv16(vp + (((ct + 1) & 15) * 16) * VPITCH);
      acc[ct] = __builtin_amdgcn_wmma_f32_16x16x32_bf16(
          false, pA, false, vf, (short)0, acc[ct], false, false);
      vf = vn;
    }
    __syncthreads();                  // done reading buf before it's refilled
  }

  // ---- finalize: y = acc / l; store Yt[b][row0+m][c] as bf16
  v8f ivec;
#pragma unroll
  for (int v = 0; v < 8; ++v) ivec[v] = 1.0f / lrun[v];
  bf16* ybase = Yt + ((size_t)b * NN + row0) * CC;
#pragma unroll
  for (int ct = 0; ct < 16; ++ct) {
    v8f o = acc[ct] * ivec;
#pragma unroll
    for (int v = 0; v < 8; ++v) {
      const int m = v + hoff;
      ybase[(size_t)m * CC + ct * 16 + l15] = (bf16)o[v];
    }
  }
}

// ---------------------------------------------------------------------------
// 1x1 conv + bias + residual. W A-fragments cached; each wave sweeps 8 n-tiles
// with double-buffered B loads.
// ---------------------------------------------------------------------------
__global__ __launch_bounds__(128, 1) void conv_kernel(
    const bf16* __restrict__ Wbf, const bf16* __restrict__ Yt,
    const float* __restrict__ Wb, const float* __restrict__ x,
    float* __restrict__ out) {
  const int lane = threadIdx.x & 31;
  const int wave = threadIdx.x >> 5;
  const int l15  = lane & 15;
  const int hsel = lane >> 4;
  const int hoff = hsel * 8;
  const int boff = hsel * 16;
  const int o0   = (blockIdx.y * 4 + wave) * 16;
  const int b    = blockIdx.z;

  v16bf wa[8];
  const bf16* wbase = Wbf + (size_t)(o0 + l15) * CC;
#pragma unroll
  for (int kt = 0; kt < 8; ++kt) {
    const bf16* p = wbase + kt * 32 + boff;
    v8bf lo = *(const v8bf*)p;
    v8bf hi = *(const v8bf*)(p + 8);
#pragma unroll
    for (int e = 0; e < 8; ++e) { wa[kt][e] = lo[e]; wa[kt][e + 8] = hi[e]; }
  }
  float bias[8];
#pragma unroll
  for (int v = 0; v < 8; ++v) bias[v] = Wb[o0 + v + hoff];

  for (int i = 0; i < 8; ++i) {
    const int n0 = (blockIdx.x * 8 + i) * 16;
    const bf16* ybase = Yt + ((size_t)b * NN + n0 + l15) * CC + boff;
    v8f acc = (v8f){};
    v16bf bf0 = *(const v16bf*)(ybase);
#pragma unroll
    for (int kt = 0; kt < 8; ++kt) {
      v16bf bn = *(const v16bf*)(ybase + (((kt + 1) & 7)) * 32);
      acc = __builtin_amdgcn_wmma_f32_16x16x32_bf16(false, wa[kt], false, bf0,
                                                    (short)0, acc, false, false);
      bf0 = bn;
    }
#pragma unroll
    for (int v = 0; v < 8; ++v) {
      const int o = o0 + v + hoff;
      const size_t idx = ((size_t)b * CC + o) * NN + n0 + l15;
      out[idx] = acc[v] + bias[v] + x[idx];
    }
  }
}

// ---------------------------------------------------------------------------
extern "C" void kernel_launch(void* const* d_in, const int* in_sizes, int n_in,
                              void* d_out, int out_size, void* d_ws,
                              size_t ws_size, hipStream_t stream) {
  const float* x  = (const float*)d_in[0];
  const float* Ww = (const float*)d_in[1];
  const float* Wb = (const float*)d_in[2];
  float* out = (float*)d_out;

  char* ws = (char*)d_ws;
  const size_t szX = (size_t)BB * CC * NN * sizeof(bf16);
  bf16* Xb  = (bf16*)ws;            ws += szX;
  bf16* Xt  = (bf16*)ws;            ws += szX;
  bf16* Yt  = (bf16*)ws;            ws += szX;
  bf16* Wbf = (bf16*)ws;

  cvt_x_kernel<<<dim3(NN / 32, CC / 32, BB), 256, 0, stream>>>(x, Xb, Xt);
  cvt_w_kernel<<<dim3((CC * CC) / 256), 256, 0, stream>>>(Ww, Wbf);
  attn_kernel<<<dim3(NN / (WAVES * 16), BB), WAVES * 32, 0, stream>>>(Xt, Xb, Yt);
  conv_kernel<<<dim3(NN / 16 / 8, CC / 64, BB), 128, 0, stream>>>(Wbf, Yt, Wb, x, out);
}
